// BlockDS_16887811408513
// MI455X (gfx1250) — compile-verified, matrix operand order checked by reading, not run on hardware
//
#include <hip/hip_runtime.h>
#include <hip/hip_bf16.h>
#include <math.h>

// ---------------------------------------------------------------------------
// Types for CDNA5 WMMA (wave32): v_wmma_f32_16x16x32_bf16
// ---------------------------------------------------------------------------
typedef __attribute__((ext_vector_type(16))) __bf16 bf16x16;
typedef __attribute__((ext_vector_type(8)))  __bf16 bf16x8;
typedef __attribute__((ext_vector_type(4)))  __bf16 bf16x4;
typedef __attribute__((ext_vector_type(8)))  float  f32x8;
typedef __attribute__((ext_vector_type(4)))  int    int4v;

// Optional CDNA5 async global->LDS path (ASYNCcnt-tracked, no VGPR round trip).
// Builtin signature (from probe): (int4 __device__* src, int4 __shared__* dst,
// imm offset, imm cpol); generic pointers convert implicitly in HIP.
#if defined(__AMDGCN__) && __has_builtin(__builtin_amdgcn_global_load_async_to_lds_b128) && \
    __has_builtin(__builtin_amdgcn_s_wait_asynccnt)
#define HAVE_ASYNC_LDS 1
#else
#define HAVE_ASYNC_LDS 0
#endif

// A/B 16-bit fragment layout (ISA 7.12.2): lane L<16 -> row L, K elements
// {koff+0..7, koff+16..23} with koff = (L>=16)?8:0. Two contiguous 16B chunks.
static __device__ __forceinline__ bf16x16 load_frag(const __bf16* p) {
  bf16x8 c0 = *(const bf16x8*)p;
  bf16x8 c1 = *(const bf16x8*)(p + 16);
  bf16x16 r;
#pragma unroll
  for (int i = 0; i < 8; ++i) { r[i] = c0[i]; r[i + 8] = c1[i]; }
  return r;
}

static __device__ __forceinline__ f32x8 wmma_bf16(bf16x16 a, bf16x16 b, f32x8 c) {
  // (neg_a, A, neg_b, B, c_mod, C, reuse_a, reuse_b)
  return __builtin_amdgcn_wmma_f32_16x16x32_bf16(false, a, false, b, (short)0, c,
                                                 false, false);
}

// ---------------------------------------------------------------------------
// fp32 -> bf16 converter (weights)
// ---------------------------------------------------------------------------
__global__ void cvt_kernel(const float* __restrict__ s, __bf16* __restrict__ d, int n) {
  int i = (blockIdx.x * 256 + threadIdx.x) * 4;
  if (i + 3 < n) {
    float4 v = *(const float4*)(s + i);
    bf16x4 o;
    o[0] = (__bf16)v.x; o[1] = (__bf16)v.y; o[2] = (__bf16)v.z; o[3] = (__bf16)v.w;
    *(bf16x4*)(d + i) = o;
  }
}

// ---------------------------------------------------------------------------
// LayerNorm over rows of 1024 fp32 -> bf16 (one block per row)
// ---------------------------------------------------------------------------
__global__ __launch_bounds__(256) void ln_kernel(const float* __restrict__ x,
                                                 const float* __restrict__ w,
                                                 const float* __restrict__ bvec,
                                                 __bf16* __restrict__ out) {
  __shared__ float rs[256];
  __shared__ float rq[256];
  const int tid = threadIdx.x;
  const size_t row = blockIdx.x;
  const float* xr = x + row * 1024;
  float4 v = *(const float4*)(xr + tid * 4);
  float vv[4] = {v.x, v.y, v.z, v.w};
  rs[tid] = vv[0] + vv[1] + vv[2] + vv[3];
  rq[tid] = vv[0] * vv[0] + vv[1] * vv[1] + vv[2] * vv[2] + vv[3] * vv[3];
  __syncthreads();
  for (int o = 128; o > 0; o >>= 1) {
    if (tid < o) { rs[tid] += rs[tid + o]; rq[tid] += rq[tid + o]; }
    __syncthreads();
  }
  float mu  = rs[0] * (1.0f / 1024.0f);
  float var = rq[0] * (1.0f / 1024.0f) - mu * mu;
  float inv = rsqrtf(var + 1e-5f);
  int c = tid * 4;
  bf16x4 ov;
#pragma unroll
  for (int i = 0; i < 4; ++i)
    ov[i] = (__bf16)((vv[i] - mu) * inv * w[c + i] + bvec[c + i]);
  *(bf16x4*)(out + row * 1024 + c) = ov;
}

// ---------------------------------------------------------------------------
// Tiled bf16 GEMM: Out[M,N] = A[M,K] @ W[N,K]^T, fused epilogues.
// Block tile 128x128, K-step 64, 8 waves each owning a 32x64 subtile
// (16 WMMAs per wave per barrier pair).
// MODE 0: qkv split-write (fp32 d_out segments + bf16 copy)
// MODE 1: + bias + residual -> fp32            (proj)
// MODE 2: + bias, exact GELU -> bf16           (fc1)
// MODE 3: + bias -> hidden_out; + residual -> x_out   (fc2)
// ---------------------------------------------------------------------------
#define BM 128
#define BN 128
#define BK 64
#define LDSS 72  // padded row stride (bf16): 144B rows, 16B aligned, no bank conflict

template <int MODE>
__global__ __launch_bounds__(256) void gemm_kernel(
    const __bf16* __restrict__ A, const __bf16* __restrict__ W,
    const float* __restrict__ bias, const float* __restrict__ resid,
    float* __restrict__ outf, float* __restrict__ outf2,
    __bf16* __restrict__ outb, int M, int N, int K) {
  __shared__ __align__(16) __bf16 As[BM * LDSS];
  __shared__ __align__(16) __bf16 Bs[BN * LDSS];
  const int tid  = threadIdx.x;
  const int lane = tid & 31;
  const int wv   = tid >> 5;
  const int wm   = (wv >> 1) * 32;  // wave row origin (0,32,64,96)
  const int wn   = (wv & 1) * 64;   // wave col origin (0,64)
  const int m0   = blockIdx.y * BM;
  const int n0   = blockIdx.x * BN;
  const int lr   = lane & 15;
  const int koff = (lane >> 4) * 8;

  f32x8 acc[2][4];
#pragma unroll
  for (int i = 0; i < 2; ++i)
#pragma unroll
    for (int j = 0; j < 4; ++j)
#pragma unroll
      for (int v = 0; v < 8; ++v) acc[i][j][v] = 0.0f;

  for (int k0 = 0; k0 < K; k0 += BK) {
    // Stage 128x64 bf16 tiles of A and W into LDS (4x 16B chunks per thread each).
#pragma unroll
    for (int i = 0; i < 4; ++i) {
      int c   = tid * 4 + i;
      int row = c >> 3;
      int cc  = (c & 7) * 8;
#if HAVE_ASYNC_LDS
      __builtin_amdgcn_global_load_async_to_lds_b128(
          (int4v*)&A[(size_t)(m0 + row) * K + (k0 + cc)],
          (int4v*)&As[row * LDSS + cc], 0, 0);
      __builtin_amdgcn_global_load_async_to_lds_b128(
          (int4v*)&W[(size_t)(n0 + row) * K + (k0 + cc)],
          (int4v*)&Bs[row * LDSS + cc], 0, 0);
#else
      *(uint4*)&As[row * LDSS + cc] = *(const uint4*)&A[(size_t)(m0 + row) * K + (k0 + cc)];
      *(uint4*)&Bs[row * LDSS + cc] = *(const uint4*)&W[(size_t)(n0 + row) * K + (k0 + cc)];
#endif
      if (k0 + BK < K) {
        __builtin_prefetch(&A[(size_t)(m0 + row) * K + (k0 + BK + cc)], 0, 3);
        __builtin_prefetch(&W[(size_t)(n0 + row) * K + (k0 + BK + cc)], 0, 3);
      }
    }
#if HAVE_ASYNC_LDS
    __builtin_amdgcn_s_wait_asynccnt(0);
#endif
    __syncthreads();
    bf16x16 af[2][2], bfr[4][2];
#pragma unroll
    for (int mt = 0; mt < 2; ++mt)
#pragma unroll
      for (int ks = 0; ks < 2; ++ks)
        af[mt][ks] = load_frag(&As[(wm + mt * 16 + lr) * LDSS + ks * 32 + koff]);
#pragma unroll
    for (int nt = 0; nt < 4; ++nt)
#pragma unroll
      for (int ks = 0; ks < 2; ++ks)
        bfr[nt][ks] = load_frag(&Bs[(wn + nt * 16 + lr) * LDSS + ks * 32 + koff]);
#pragma unroll
    for (int mt = 0; mt < 2; ++mt)
#pragma unroll
      for (int nt = 0; nt < 4; ++nt)
#pragma unroll
        for (int ks = 0; ks < 2; ++ks)
          acc[mt][nt] = wmma_bf16(af[mt][ks], bfr[nt][ks], acc[mt][nt]);
    __syncthreads();
  }

  // Epilogue. C/D layout: lane<16 -> N=lane, M=vgpr; lane>=16 -> N=lane-16, M=vgpr+8.
  const int hi = (lane >> 4) * 8;
#pragma unroll
  for (int mt = 0; mt < 2; ++mt) {
#pragma unroll
    for (int nt = 0; nt < 4; ++nt) {
      int n = n0 + wn + nt * 16 + lr;
#pragma unroll
      for (int v = 0; v < 8; ++v) {
        int m = m0 + wm + mt * 16 + hi + v;
        float val = acc[mt][nt][v];
        if (MODE == 0) {
          int seg = n >> 10, c2 = n & 1023;
          outf[(size_t)seg * 8388608 + (size_t)m * 1024 + c2] = val;  // q/k/v fp32
          outb[(size_t)m * N + n] = (__bf16)val;                      // bf16 for attn
        } else if (MODE == 1) {
          outf[(size_t)m * N + n] = val + bias[n] + resid[(size_t)m * N + n];
        } else if (MODE == 2) {
          float t = val + bias[n];
          outb[(size_t)m * N + n] =
              (__bf16)(0.5f * t * (1.0f + erff(t * 0.70710678118654752f)));
        } else {
          float hv = val + bias[n];
          outf[(size_t)m * N + n]  = hv;                               // hidden_out
          outf2[(size_t)m * N + n] = hv + resid[(size_t)m * N + n];    // final x
        }
      }
    }
  }
}

// ---------------------------------------------------------------------------
// Flash attention: one wave per (b, h, 32-query block), online softmax over
// 32-key tiles. hd=64, scale=0.125. qkv bf16 layout: [8192 rows, 3072 cols].
// ---------------------------------------------------------------------------
__global__ __launch_bounds__(32) void attn_kernel(const __bf16* __restrict__ qkv,
                                                  __bf16* __restrict__ ao) {
  __shared__ __align__(16) float  Sx[32][33];
  __shared__ __align__(16) __bf16 P[32][40];
  __shared__ __align__(16) __bf16 Vt[64][40];
  __shared__ float cf[32];

  const int lane = threadIdx.x;
  const int q0   = blockIdx.x * 32;
  const int h    = blockIdx.y;
  const int b    = blockIdx.z;
  const int lr   = lane & 15;
  const int koff = (lane >> 4) * 8;
  const int hi   = koff;

  const __bf16* Qb = qkv + ((size_t)b * 1024) * 3072 + h * 64;
  const __bf16* Kb = Qb + 1024;
  const __bf16* Vb = Qb + 2048;

  // Q fragments resident for the whole pass: [mt: query half][ks: d half]
  bf16x16 qf[2][2];
#pragma unroll
  for (int mt = 0; mt < 2; ++mt)
#pragma unroll
    for (int ks = 0; ks < 2; ++ks)
      qf[mt][ks] = load_frag(Qb + (size_t)(q0 + mt * 16 + lr) * 3072 + ks * 32 + koff);

  f32x8 o[2][4];
#pragma unroll
  for (int i = 0; i < 2; ++i)
#pragma unroll
    for (int j = 0; j < 4; ++j)
#pragma unroll
      for (int v = 0; v < 8; ++v) o[i][j][v] = 0.0f;
  float mrow = -INFINITY, lrow = 0.0f;  // lane L owns softmax state of query q0+L

  for (int kb = 0; kb < 1024; kb += 32) {
    // S = Q @ K^T (K-dim = 64 -> two WMMAs per 16x16 tile)
    bf16x16 kf[2][2];
#pragma unroll
    for (int nt = 0; nt < 2; ++nt)
#pragma unroll
      for (int ks = 0; ks < 2; ++ks)
        kf[nt][ks] = load_frag(Kb + (size_t)(kb + nt * 16 + lr) * 3072 + ks * 32 + koff);
    f32x8 sa[2][2];
#pragma unroll
    for (int i = 0; i < 2; ++i)
#pragma unroll
      for (int j = 0; j < 2; ++j)
#pragma unroll
        for (int v = 0; v < 8; ++v) sa[i][j][v] = 0.0f;
#pragma unroll
    for (int mt = 0; mt < 2; ++mt)
#pragma unroll
      for (int nt = 0; nt < 2; ++nt)
#pragma unroll
        for (int ks = 0; ks < 2; ++ks)
          sa[mt][nt] = wmma_bf16(qf[mt][ks], kf[nt][ks], sa[mt][nt]);
#pragma unroll
    for (int mt = 0; mt < 2; ++mt)
#pragma unroll
      for (int nt = 0; nt < 2; ++nt)
#pragma unroll
        for (int v = 0; v < 8; ++v)
          Sx[mt * 16 + hi + v][nt * 16 + lr] = sa[mt][nt][v] * 0.125f;
    __syncthreads();

    // Online softmax: lane L processes query row L.
    float rmax = -INFINITY;
#pragma unroll
    for (int j = 0; j < 32; ++j) rmax = fmaxf(rmax, Sx[lane][j]);
    float mnew = fmaxf(mrow, rmax);
    float corr = __expf(mrow - mnew);
    float ps = 0.0f;
#pragma unroll
    for (int j = 0; j < 32; ++j) {
      float p = __expf(Sx[lane][j] - mnew);
      ps += p;
      P[lane][j] = (__bf16)p;
    }
    lrow = lrow * corr + ps;
    mrow = mnew;
    cf[lane] = corr;

    // Stage V tile transposed: Vt[d][key] so P@V B-fragments are contiguous.
    const __bf16* vr = Vb + (size_t)(kb + lane) * 3072;
#pragma unroll
    for (int d0 = 0; d0 < 64; d0 += 8) {
      bf16x8 vv = *(const bf16x8*)(vr + d0);
#pragma unroll
      for (int i = 0; i < 8; ++i) Vt[d0 + i][lane] = vv[i];
    }
    __syncthreads();

    // Rescale running O by per-row correction.
#pragma unroll
    for (int mt = 0; mt < 2; ++mt)
#pragma unroll
      for (int v = 0; v < 8; ++v) {
        float c = cf[mt * 16 + hi + v];
#pragma unroll
        for (int dt = 0; dt < 4; ++dt) o[mt][dt][v] *= c;
      }

    // O += P @ V
    bf16x16 pf[2];
#pragma unroll
    for (int mt = 0; mt < 2; ++mt)
      pf[mt] = load_frag(&P[mt * 16 + lr][koff]);
    bf16x16 vf[4];
#pragma unroll
    for (int dt = 0; dt < 4; ++dt)
      vf[dt] = load_frag(&Vt[dt * 16 + lr][koff]);
#pragma unroll
    for (int mt = 0; mt < 2; ++mt)
#pragma unroll
      for (int dt = 0; dt < 4; ++dt)
        o[mt][dt] = wmma_bf16(pf[mt], vf[dt], o[mt][dt]);
    __syncthreads();
  }

  cf[lane] = 1.0f / lrow;
  __syncthreads();
#pragma unroll
  for (int mt = 0; mt < 2; ++mt)
#pragma unroll
    for (int v = 0; v < 8; ++v) {
      float li = cf[mt * 16 + hi + v];
      int r = q0 + mt * 16 + hi + v;
#pragma unroll
      for (int dt = 0; dt < 4; ++dt) {
        int col = h * 64 + dt * 16 + lr;
        ao[((size_t)b * 1024 + r) * 1024 + col] = (__bf16)(o[mt][dt][v] * li);
      }
    }
}

// ---------------------------------------------------------------------------
// Host orchestration
// ---------------------------------------------------------------------------
extern "C" void kernel_launch(void* const* d_in, const int* in_sizes, int n_in,
                              void* d_out, int out_size, void* d_ws, size_t ws_size,
                              hipStream_t stream) {
  (void)in_sizes; (void)n_in; (void)out_size; (void)ws_size;
  const float* x      = (const float*)d_in[0];
  const float* n1w    = (const float*)d_in[1];
  const float* n1b    = (const float*)d_in[2];
  const float* qkv_w  = (const float*)d_in[3];
  const float* proj_w = (const float*)d_in[4];
  const float* proj_b = (const float*)d_in[5];
  const float* n2w    = (const float*)d_in[6];
  const float* n2b    = (const float*)d_in[7];
  const float* fc1_w  = (const float*)d_in[8];
  const float* fc1_b  = (const float*)d_in[9];
  const float* fc2_w  = (const float*)d_in[10];
  const float* fc2_b  = (const float*)d_in[11];
  float* out = (float*)d_out;

  char* ws = (char*)d_ws;
  size_t off = 0;
  auto take = [&](size_t bytes) -> char* {
    char* p = ws + off;
    off = (off + bytes + 255) & ~(size_t)255;
    return p;
  };
  __bf16* wq   = (__bf16*)take(3072ull * 1024 * 2);
  __bf16* wp   = (__bf16*)take(1024ull * 1024 * 2);
  __bf16* w1   = (__bf16*)take(4096ull * 1024 * 2);
  __bf16* w2   = (__bf16*)take(1024ull * 4096 * 2);
  __bf16* xn   = (__bf16*)take(8192ull * 1024 * 2);  // reused for both LayerNorms
  __bf16* qkvb = (__bf16*)take(8192ull * 3072 * 2);
  __bf16* aob  = (__bf16*)take(8192ull * 1024 * 2);
  float*  x1   = (float*)take(8192ull * 1024 * 4);
  __bf16* hbuf = (__bf16*)take(8192ull * 4096 * 2);

  const size_t SEG = 8388608;  // 8*1024*1024

  // Weights -> bf16
  cvt_kernel<<<3072, 256, 0, stream>>>(qkv_w,  wq, 3072 * 1024);
  cvt_kernel<<<1024, 256, 0, stream>>>(proj_w, wp, 1024 * 1024);
  cvt_kernel<<<4096, 256, 0, stream>>>(fc1_w,  w1, 4096 * 1024);
  cvt_kernel<<<4096, 256, 0, stream>>>(fc2_w,  w2, 1024 * 4096);

  // Attention branch
  ln_kernel<<<8192, 256, 0, stream>>>(x, n1w, n1b, xn);
  gemm_kernel<0><<<dim3(3072 / BN, 8192 / BM), 256, 0, stream>>>(
      xn, wq, nullptr, nullptr, out + SEG, nullptr, qkvb, 8192, 3072, 1024);
  attn_kernel<<<dim3(32, 16, 8), 32, 0, stream>>>(qkvb, aob);
  gemm_kernel<1><<<dim3(1024 / BN, 8192 / BM), 256, 0, stream>>>(
      aob, wp, proj_b, x, x1, nullptr, nullptr, 8192, 1024, 1024);

  // MLP branch
  ln_kernel<<<8192, 256, 0, stream>>>(x1, n2w, n2b, xn);
  gemm_kernel<2><<<dim3(4096 / BN, 8192 / BM), 256, 0, stream>>>(
      xn, w1, fc1_b, nullptr, nullptr, nullptr, hbuf, 8192, 4096, 1024);
  gemm_kernel<3><<<dim3(1024 / BN, 8192 / BM), 256, 0, stream>>>(
      hbuf, w2, fc2_b, x1, out + 4 * SEG, out, nullptr, 8192, 1024, 4096);
}
